// RNNDecoder_51170240364896
// MI455X (gfx1250) — matrix operand order repaired and checked by reading, not compiled, not run on hardware
//
#include <hip/hip_runtime.h>
#include <hip/hip_bf16.h>
#include <math.h>

// ---------------------------------------------------------------------------
// MI455X (gfx1250) LSTM greedy decoder.
// Strategy: convert all weights to bf16 (transposed to [K x N]) once per
// launch -> ~78MB total, resident in the 192MB L2 for all 64 serial steps.
// All GEMMs use v_wmma_f32_16x16x32_bf16 (wave32, 16x16 tiles, fp32 accum).
// ---------------------------------------------------------------------------

#define DI __device__ __forceinline__

typedef __attribute__((ext_vector_type(16))) __bf16 bf16x16;
typedef __attribute__((ext_vector_type(8)))  float  f32x8;

static constexpr int Vv = 32000, Ee = 512, Hh = 1024, IHih = 1024, Bb = 32, Tt = 64;
static constexpr int G4 = 4 * Hh;            // 4096 gates
static constexpr int PROJ_BLOCKS = Vv / 128; // 250

struct alignas(16) U4 { unsigned int x, y, z, w; };
union FragU { bf16x16 v; U4 q[2]; };

DI unsigned short f2bf(float f) {              // round-to-nearest-even fp32->bf16
  unsigned int x = __float_as_uint(f);
  x += 0x7fffu + ((x >> 16) & 1u);
  return (unsigned short)(x >> 16);
}

DI f32x8 wmma_bf16(bf16x16 a, bf16x16 b, f32x8 c) {
  // (neg_a, A, neg_b, B, c_mod, C, reuse_a, reuse_b)
  return __builtin_amdgcn_wmma_f32_16x16x32_bf16(false, a, false, b, (short)0, c,
                                                 false, false);
}

// A: row-major [32 x K] bf16. Fragment = rows m0..m0+15, cols k0..k0+31.
// gfx1250 16-bit A layout: lanes 0-15 -> M=lane, elems 0-7:K=k0..k0+7,
// elems 8-15:K=k0+16..k0+23; lanes 16-31 shift K by +8.  Two 16B loads/lane.
DI bf16x16 load_a(const unsigned short* A, int K, int m0, int k0, int lane) {
  int r = lane & 15, hh = lane >> 4;
  const unsigned short* p = A + (m0 + r) * K + k0 + 8 * hh;
  FragU f;
  f.q[0] = *(const U4*)(p);
  f.q[1] = *(const U4*)(p + 16);
  return f.v;
}
// B: row-major [K x N] bf16. Fragment = rows k0..k0+31, cols n0..n0+15.
// gfx1250 16-bit B layout: lanes 0-15 -> N=lane, elems 0-7:K=k0..k0+7,
// elems 8-15:K=k0+16..k0+23; lanes 16-31 shift K by +8. Strided scalar loads.
DI bf16x16 load_b(const unsigned short* Bm, int N, int k0, int n0, int lane) {
  int c = lane & 15, hh = lane >> 4;
  const unsigned short* p = Bm + (size_t)(k0 + 8 * hh) * N + n0 + c;
  FragU f;
  unsigned short* e = (unsigned short*)&f;
  for (int s = 0; s < 8; ++s) {
    e[s] = p[(size_t)s * N];
    e[8 + s] = p[(size_t)(16 + s) * N];
  }
  return f.v;
}

DI float sigm(float x) { return 1.0f / (1.0f + expf(-x)); }

// --------------------------- prep kernels ----------------------------------

// dst[k*N + n] = bf16(src[n*K + k])  (weight transpose + convert, one-time)
__global__ void k_convT(const float* __restrict__ src,
                        unsigned short* __restrict__ dst, int N, int K) {
  int idx = blockIdx.x * 256 + threadIdx.x;
  if (idx < N * K) {
    int n = idx / K, k = idx - n * K;
    dst[(size_t)k * N + n] = f2bf(src[idx]);
  }
}

__global__ void k_prep(const float* __restrict__ b_ih, const float* __restrict__ b_hh,
                       float* __restrict__ bias, const float* __restrict__ init_hidden,
                       unsigned short* __restrict__ xbf,
                       unsigned short* __restrict__ prevbf) {
  int idx = blockIdx.x * 256 + threadIdx.x;
  if (idx < G4) bias[idx] = b_ih[idx] + b_hh[idx];
  if (idx < Bb * IHih) xbf[idx] = f2bf(init_hidden[idx]);
  if (idx < Bb * Ee) prevbf[idx] = 0;  // bf16(0)
}

// --------------------------- bridge GEMM -----------------------------------
// h0 = tanh(init_hidden @ bridge_w.T + bridge_b);  c0 = h0
__global__ void __launch_bounds__(256)
k_bridge(const unsigned short* __restrict__ xbf, const unsigned short* __restrict__ brT,
         const float* __restrict__ bridge_b, float* __restrict__ hf,
         unsigned short* __restrict__ hbf, float* __restrict__ cf) {
  int wid = threadIdx.x >> 5, lane = threadIdx.x & 31;
  int n0 = blockIdx.x * 128 + wid * 16;
  f32x8 acc0 = {0, 0, 0, 0, 0, 0, 0, 0}, acc1 = acc0;
  for (int k0 = 0; k0 < IHih; k0 += 32) {
    bf16x16 b  = load_b(brT, Hh, k0, n0, lane);
    bf16x16 a0 = load_a(xbf, IHih, 0,  k0, lane);
    bf16x16 a1 = load_a(xbf, IHih, 16, k0, lane);
    acc0 = wmma_bf16(a0, b, acc0);
    acc1 = wmma_bf16(a1, b, acc1);
  }
  int n = n0 + (lane & 15);
  float bb = bridge_b[n];
  int mh = (lane >> 4) * 8;
  for (int r = 0; r < 8; ++r) {
    int m0 = r + mh, m1 = m0 + 16;
    float v0 = tanhf(acc0[r] + bb);
    float v1 = tanhf(acc1[r] + bb);
    hf[m0 * Hh + n] = v0; cf[m0 * Hh + n] = v0; hbf[m0 * Hh + n] = f2bf(v0);
    hf[m1 * Hh + n] = v1; cf[m1 * Hh + n] = v1; hbf[m1 * Hh + n] = f2bf(v1);
  }
}

// ----------------- fused gates GEMM + LSTM pointwise -----------------------
// Block owns 32 hidden units x all 4 gates: 8 waves, one 32x16 WMMA tile each.
__global__ void __launch_bounds__(256)
k_step(const unsigned short* __restrict__ prevbf, const unsigned short* __restrict__ hbf_in,
       const unsigned short* __restrict__ wihT, const unsigned short* __restrict__ whhT,
       const float* __restrict__ bias, float* __restrict__ cf, float* __restrict__ hf,
       unsigned short* __restrict__ hbf_out, float* __restrict__ out_t) {
  __shared__ float g[4][32][33];  // [gate][batch][hidden-local]
  int wid = threadIdx.x >> 5, lane = threadIdx.x & 31;
  int gi = wid >> 1, s = wid & 1;
  int j0 = blockIdx.x * 32;
  int n0 = gi * Hh + j0 + s * 16;  // column in 4096-wide gate space
  f32x8 acc0 = {0, 0, 0, 0, 0, 0, 0, 0}, acc1 = acc0;
  for (int k0 = 0; k0 < Ee; k0 += 32) {         // prev @ w_ih.T
    bf16x16 b  = load_b(wihT, G4, k0, n0, lane);
    bf16x16 a0 = load_a(prevbf, Ee, 0,  k0, lane);
    bf16x16 a1 = load_a(prevbf, Ee, 16, k0, lane);
    acc0 = wmma_bf16(a0, b, acc0);
    acc1 = wmma_bf16(a1, b, acc1);
  }
  for (int k0 = 0; k0 < Hh; k0 += 32) {         // h @ w_hh.T
    bf16x16 b  = load_b(whhT, G4, k0, n0, lane);
    bf16x16 a0 = load_a(hbf_in, Hh, 0,  k0, lane);
    bf16x16 a1 = load_a(hbf_in, Hh, 16, k0, lane);
    acc0 = wmma_bf16(a0, b, acc0);
    acc1 = wmma_bf16(a1, b, acc1);
  }
  float bv = bias[n0 + (lane & 15)];
  int col = s * 16 + (lane & 15);
  int mh = (lane >> 4) * 8;
  for (int r = 0; r < 8; ++r) {
    g[gi][r + mh][col]      = acc0[r] + bv;
    g[gi][r + mh + 16][col] = acc1[r] + bv;
  }
  __syncthreads();
  for (int it = 0; it < 4; ++it) {
    int idx = threadIdx.x + it * 256;
    int b = idx >> 5, jl = idx & 31;
    int j = j0 + jl;
    float iv = sigm(g[0][b][jl]);
    float fv = sigm(g[1][b][jl]);
    float gv = tanhf(g[2][b][jl]);
    float ov = sigm(g[3][b][jl]);
    float cn = fv * cf[b * Hh + j] + iv * gv;
    float hn = ov * tanhf(cn);
    cf[b * Hh + j] = cn;
    hf[b * Hh + j] = hn;
    hbf_out[b * Hh + j] = f2bf(hn);
    out_t[b * Hh + j] = hn;
  }
}

// --------------------------- projection GEMM -------------------------------
// logits = h_new @ proj_w.T + proj_b ; write logits + per-block max/argmax.
__global__ void __launch_bounds__(256)
k_proj(const unsigned short* __restrict__ hbf, const unsigned short* __restrict__ projT,
       const float* __restrict__ proj_b, float* __restrict__ logits_t,
       float* __restrict__ pmax, int* __restrict__ pidx) {
  __shared__ float lt[32][129];
  int wid = threadIdx.x >> 5, lane = threadIdx.x & 31;
  int n0blk = blockIdx.x * 128;
  int n0 = n0blk + wid * 16;
  f32x8 acc0 = {0, 0, 0, 0, 0, 0, 0, 0}, acc1 = acc0;
  for (int k0 = 0; k0 < Hh; k0 += 32) {
    bf16x16 b  = load_b(projT, Vv, k0, n0, lane);
    bf16x16 a0 = load_a(hbf, Hh, 0,  k0, lane);
    bf16x16 a1 = load_a(hbf, Hh, 16, k0, lane);
    acc0 = wmma_bf16(a0, b, acc0);
    acc1 = wmma_bf16(a1, b, acc1);
  }
  float pb = proj_b[n0 + (lane & 15)];
  int col = wid * 16 + (lane & 15);
  int mh = (lane >> 4) * 8;
  for (int r = 0; r < 8; ++r) {
    lt[r + mh][col]      = acc0[r] + pb;
    lt[r + mh + 16][col] = acc1[r] + pb;
  }
  __syncthreads();
  for (int it = 0; it < 16; ++it) {  // coalesced store of 32x128 tile
    int idx = threadIdx.x + it * 256;
    int m = idx >> 7, n = idx & 127;
    logits_t[(size_t)m * Vv + n0blk + n] = lt[m][n];
  }
  if (threadIdx.x < 32) {            // per-row block max + first argmax
    int m = threadIdx.x;
    float mx = lt[m][0]; int am = 0;
    for (int n = 1; n < 128; ++n) {
      float vv = lt[m][n];
      if (vv > mx) { mx = vv; am = n; }
    }
    pmax[blockIdx.x * 32 + m] = mx;
    pidx[blockIdx.x * 32 + m] = n0blk + am;
  }
}

// ---------------- log-softmax finalize + argmax gather ---------------------
__global__ void __launch_bounds__(256)
k_final(float* __restrict__ row_base, const float* __restrict__ emb,
        unsigned short* __restrict__ prevbf, const float* __restrict__ pmax,
        const int* __restrict__ pidx) {
  __shared__ float sv[256];
  __shared__ int si[256];
  __shared__ float s_max, s_lse;
  __shared__ int s_idx;
  int b = blockIdx.x, tid = threadIdx.x;
  float v = -3.402823466e38f;
  int ii = 0x7fffffff;
  if (tid < PROJ_BLOCKS) { v = pmax[tid * 32 + b]; ii = pidx[tid * 32 + b]; }
  sv[tid] = v; si[tid] = ii;
  __syncthreads();
  for (int s = 128; s > 0; s >>= 1) {  // max + lowest-index tie break
    if (tid < s) {
      float v2 = sv[tid + s]; int i2 = si[tid + s];
      if (v2 > sv[tid] || (v2 == sv[tid] && i2 < si[tid])) { sv[tid] = v2; si[tid] = i2; }
    }
    __syncthreads();
  }
  if (tid == 0) { s_max = sv[0]; s_idx = si[0]; }
  __syncthreads();
  float rowmax = s_max;
  float* row = row_base + (size_t)b * Vv;
  float acc = 0.f;
  for (int x = tid; x < Vv; x += 256) acc += expf(row[x] - rowmax);
  __syncthreads();
  sv[tid] = acc;
  __syncthreads();
  for (int s = 128; s > 0; s >>= 1) {
    if (tid < s) sv[tid] += sv[tid + s];
    __syncthreads();
  }
  if (tid == 0) s_lse = rowmax + logf(sv[0]);
  __syncthreads();
  float lse = s_lse;
  for (int x = tid; x < Vv; x += 256) row[x] = row[x] - lse;  // logp in place
  int widx = s_idx;
  for (int e = tid; e < Ee; e += 256)  // prev = bf16(emb_table[argmax])
    prevbf[b * Ee + e] = f2bf(emb[(size_t)widx * Ee + e]);
}

__global__ void k_hidden(const float* __restrict__ hf, const float* __restrict__ cf,
                         float* __restrict__ out) {
  int idx = blockIdx.x * 256 + threadIdx.x;
  if (idx < Bb * Hh) { out[idx] = hf[idx]; out[Bb * Hh + idx] = cf[idx]; }
}

// ---------------------------------------------------------------------------

extern "C" void kernel_launch(void* const* d_in, const int* in_sizes, int n_in,
                              void* d_out, int out_size, void* d_ws, size_t ws_size,
                              hipStream_t stream) {
  (void)in_sizes; (void)n_in; (void)out_size; (void)ws_size;
  // inputs (setup_inputs order): 0 trg_embed(unused), 1 init_hidden, 2 emb_table,
  // 3 w_ih, 4 w_hh, 5 b_ih, 6 b_hh, 7 bridge_w, 8 bridge_b, 9 proj_w, 10 proj_b,
  // 11 max_len(unused, ==64)
  const float* init_hidden = (const float*)d_in[1];
  const float* emb_table   = (const float*)d_in[2];
  const float* w_ih        = (const float*)d_in[3];
  const float* w_hh        = (const float*)d_in[4];
  const float* b_ih        = (const float*)d_in[5];
  const float* b_hh        = (const float*)d_in[6];
  const float* bridge_w    = (const float*)d_in[7];
  const float* bridge_b    = (const float*)d_in[8];
  const float* proj_w      = (const float*)d_in[9];
  const float* proj_b      = (const float*)d_in[10];

  float* out     = (float*)d_out;
  float* outputs = out;                                // [T,B,H]
  float* hidden  = out + (size_t)Tt * Bb * Hh;         // [2,B,H] (h then c)
  float* proj    = hidden + 2 * Bb * Hh;               // [T,B,V]

  // workspace carve-out (~80 MB total)
  char* w = (char*)d_ws;
  auto take = [&](size_t bytes) -> void* {
    void* p = (void*)w;
    w += (bytes + 255) & ~(size_t)255;
    return p;
  };
  unsigned short* wihT   = (unsigned short*)take(2ull * Ee * G4);      // [E x 4H]
  unsigned short* whhT   = (unsigned short*)take(2ull * Hh * G4);      // [H x 4H]
  unsigned short* projT  = (unsigned short*)take(2ull * Hh * Vv);      // [H x V]
  unsigned short* brT    = (unsigned short*)take(2ull * IHih * Hh);    // [IH x H]
  float*          bias   = (float*)take(4ull * G4);
  unsigned short* xbf    = (unsigned short*)take(2ull * Bb * IHih);
  unsigned short* hbf0   = (unsigned short*)take(2ull * Bb * Hh);
  unsigned short* hbf1   = (unsigned short*)take(2ull * Bb * Hh);
  unsigned short* prevbf = (unsigned short*)take(2ull * Bb * Ee);
  float*          hf     = (float*)take(4ull * Bb * Hh);
  float*          cf     = (float*)take(4ull * Bb * Hh);
  float*          pmax   = (float*)take(4ull * PROJ_BLOCKS * 32);
  int*            pidx   = (int*)take(4ull * PROJ_BLOCKS * 32);

  // one-time bf16 weight conversion (all resident in 192MB L2 afterwards)
  k_convT<<<(G4 * Ee + 255) / 256, 256, 0, stream>>>(w_ih, wihT, G4, Ee);
  k_convT<<<(G4 * Hh + 255) / 256, 256, 0, stream>>>(w_hh, whhT, G4, Hh);
  k_convT<<<(Vv * Hh + 255) / 256, 256, 0, stream>>>(proj_w, projT, Vv, Hh);
  k_convT<<<(Hh * IHih + 255) / 256, 256, 0, stream>>>(bridge_w, brT, Hh, IHih);
  k_prep<<<(Bb * IHih + 255) / 256, 256, 0, stream>>>(b_ih, b_hh, bias, init_hidden,
                                                      xbf, prevbf);
  k_bridge<<<Hh / 128, 256, 0, stream>>>(xbf, brT, bridge_b, hf, hbf0, cf);

  unsigned short* hbufs[2] = {hbf0, hbf1};
  for (int t = 0; t < Tt; ++t) {
    unsigned short* hin  = hbufs[t & 1];
    unsigned short* hout = hbufs[(t + 1) & 1];  // double-buffer h (bf16) across steps
    k_step<<<Hh / 32, 256, 0, stream>>>(prevbf, hin, wihT, whhT, bias, cf, hf, hout,
                                        outputs + (size_t)t * Bb * Hh);
    k_proj<<<PROJ_BLOCKS, 256, 0, stream>>>(hout, projT, proj_b,
                                            proj + (size_t)t * Bb * Vv, pmax, pidx);
    k_final<<<Bb, 256, 0, stream>>>(proj + (size_t)t * Bb * Vv, emb_table, prevbf,
                                    pmax, pidx);
  }
  k_hidden<<<(Bb * Hh + 255) / 256, 256, 0, stream>>>(hf, cf, hidden);
}
